// AutoFormer_78889959293178
// MI455X (gfx1250) — matrix operand order repaired and checked by reading, not compiled, not run on hardware
//
// Autoformer forward for MI455X (gfx1250, wave32, WMMA).
//
// Design notes (compile-only; reasoned from MI455X specs):
//  * ~290 GFLOP of GEMM vs ~300MB activation traffic -> memory floor ~15-30us
//    at 23.3 TB/s; all matmuls go through v_wmma_f32_16x16x32_bf16 (bf16 in,
//    fp32 accumulate). Weights pre-converted to bf16 [N,K] ("Bt") layout so
//    both A and B WMMA fragments are contiguous 16B/32B loads that match the
//    ISA 16-bit A/B VGPR striping (CDNA5 ISA 7.12.2).
//  * FFT autocorrelation replaced by the mathematically identical
//    S = Q K^T (batched WMMA GEMM) + mod-L diagonal reduction:
//    irfft(qf*conj(kf))[tau] = sum_l q[(l+tau)%L] k[l].
//  * conv1d(kernel=3, replicate pad) = 3 accumulating GEMM taps with a
//    per-batch clamped time shift on the A-row mapping (no im2col).
//  * Weights (~34MB bf16) are L2-resident (192MB L2), so direct global
//    fragment loads are fine; __builtin_prefetch -> global_prefetch_b8.
//
// Input flattening assumption: setup_inputs() dict insertion order, nested
// dicts recursively: x, w_emb, b_emb, enc{ac{wq,wk,wv,wo},conv1,conv2,ln_g,
// ln_b}, dec{ac1{..},ac2{..},lin1,lin2,lin3,conv1,conv2,lin_season,ln_g,ln_b},
// w_final, b_final  (29 tensors).

#include <hip/hip_runtime.h>

typedef __bf16 bfloat;
typedef __bf16 v8bf  __attribute__((ext_vector_type(8)));
typedef __bf16 v16bf __attribute__((ext_vector_type(16)));
typedef float  v8f   __attribute__((ext_vector_type(8)));

#define GF_ACC   1
#define GF_LEAKY 2

__device__ __forceinline__ bfloat f2bf(float f) {
    unsigned int u = __float_as_uint(f);
    unsigned int r = u + 0x7FFFu + ((u >> 16) & 1u);   // round-to-nearest-even
    unsigned short h = (unsigned short)(r >> 16);
    union { unsigned short s; bfloat b; } cv; cv.s = h; return cv.b;
}

// ---------------------------------------------------------------------------
// WMMA GEMM: C[M,N] = act( A[M,K] @ Bt[N,K]^T (+bias) (+Cprev if ACC) ) (+resid)
// A rows optionally remapped as l -> clamp(l+shift) within each L-row batch
// (replicate-padded conv taps). grid.z = batch with element strides sA/sB/sC.
// Block = 256 thr = 8 waves, wave tile 16x16, block tile 64(M) x 32(N).
// ---------------------------------------------------------------------------
__global__ __launch_bounds__(256)
void k_wmma_gemm(const bfloat* __restrict__ A, const bfloat* __restrict__ Bt,
                 float* __restrict__ C, const float* __restrict__ bias,
                 const float* __restrict__ resid,
                 int M, int N, int K, int Lrow, int shift, int flags,
                 long sA, long sB, long sC)
{
    const int z = blockIdx.z;
    A  += (long)z * sA;
    Bt += (long)z * sB;
    C  += (long)z * sC;

    const int wave = threadIdx.x >> 5;
    const int lane = threadIdx.x & 31;
    const int hsel = lane >> 4;        // half-wave select
    const int l15  = lane & 15;
    const int tileM = blockIdx.y * 64 + (wave & 3) * 16;
    const int tileN = blockIdx.x * 32 + (wave >> 2) * 16;
    if (tileM >= M || tileN >= N) return;

    // A row for this lane, with per-batch clamped shift (conv taps)
    int m  = tileM + l15;
    int bb = m / Lrow;
    int l  = m - bb * Lrow;
    int ls = l + shift;
    ls = ls < 0 ? 0 : (ls >= Lrow ? Lrow - 1 : ls);
    const bfloat* pA = A + (long)(bb * Lrow + ls) * K;
    const bfloat* pB = Bt + (long)(tileN + l15) * K;

    const int aoff = 8 * hsel;    // A: lanes0-15 K=[0..7],[16..23]; lanes16-31 +8
    const int boff = 16 * hsel;   // unused; B loads use aoff, same k order as A

    v8f acc = {};
    for (int k0 = 0; k0 < K; k0 += 32) {
        __builtin_prefetch(pA + k0 + 64, 0, 1);
        __builtin_prefetch(pB + k0 + 64, 0, 1);
        v8bf  a0 = *(const v8bf *)(pA + k0 + aoff);
        v8bf  a1 = *(const v8bf *)(pA + k0 + 16 + aoff);
        v8bf  b0 = *(const v8bf *)(pB + k0 + aoff);
        v8bf  b1 = *(const v8bf *)(pB + k0 + 16 + aoff);
        v16bf af, bf;
#pragma unroll
        for (int i = 0; i < 8; ++i) { af[i] = a0[i]; af[8 + i] = a1[i]; bf[i] = b0[i]; bf[8 + i] = b1[i]; }
        acc = __builtin_amdgcn_wmma_f32_16x16x32_bf16(
                  false, af, false, bf, (short)0, acc, false, false);
    }

#pragma unroll
    for (int r = 0; r < 8; ++r) {      // C layout: VGPR r -> M = r + 8*hsel
        int row = tileM + r + 8 * hsel;
        int col = tileN + l15;
        long idx = (long)row * N + col;
        float v = acc[r];
        if (bias)             v += bias[col];
        if (flags & GF_ACC)   v += C[idx];
        if (flags & GF_LEAKY) v = v >= 0.f ? v : 0.01f * v;
        if (resid)            v += resid[idx];
        C[idx] = v;
    }
}

// ---------------------------------------------------------------------------
// Elementwise / prep kernels
// ---------------------------------------------------------------------------
__global__ void k_f32_to_bf16(const float* __restrict__ s, bfloat* __restrict__ d, long n) {
    long i = (long)blockIdx.x * blockDim.x + threadIdx.x;
    if (i < n) d[i] = f2bf(s[i]);
}

// Wt[n*K + k] = bf16(W[k*N + n])  (weight transpose+convert, [K,N]->[N,K])
__global__ void k_transpose_bf16(const float* __restrict__ W, bfloat* __restrict__ Wt,
                                 int K, int N) {
    long i = (long)blockIdx.x * blockDim.x + threadIdx.x;
    if (i >= (long)K * N) return;
    int k = (int)(i / N);
    int n = (int)(i % N);
    Wt[(long)n * K + k] = f2bf(W[i]);
}

// conv weight W[O][I][3] -> 3 tap-major bf16 matrices Wt[t][O*I] (already [N=O,K=I])
__global__ void k_convw_tap(const float* __restrict__ W, bfloat* __restrict__ Wt,
                            int O, int I) {
    long i = (long)blockIdx.x * blockDim.x + threadIdx.x;
    long n = 3L * O * I;
    if (i >= n) return;
    int  t    = (int)(i % 3);
    long rest = i / 3;                     // = o*I + ii
    Wt[(long)t * O * I + rest] = f2bf(W[i]);
}

// series_decomp: trend = moving_avg(x, 25) with replicate pad; season = x - trend
__global__ void k_decomp(const float* __restrict__ x, float* __restrict__ season,
                         float* __restrict__ trend, int Bn, int Ln, int Cn) {
    long i = (long)blockIdx.x * blockDim.x + threadIdx.x;
    if (i >= (long)Bn * Ln * Cn) return;
    int  c = (int)(i % Cn);
    long t = i / Cn;
    int  l = (int)(t % Ln);
    int  b = (int)(t / Ln);
    const float* xb = x + (long)b * Ln * Cn + c;
    float s = 0.f;
#pragma unroll
    for (int d = -12; d <= 12; ++d) {
        int j = l + d;
        j = j < 0 ? 0 : (j >= Ln ? Ln - 1 : j);
        s += xb[(long)j * Cn];
    }
    float tr = s * (1.0f / 25.0f);
    trend[i]  = tr;
    season[i] = x[i] - tr;
}

__global__ void k_mean_l(const float* __restrict__ x, float* __restrict__ out,
                         int Bn, int Ln, int Cn) {
    int i = blockIdx.x * blockDim.x + threadIdx.x;
    if (i >= Bn * Cn) return;
    int b = i / Cn, c = i % Cn;
    const float* p = x + (long)b * Ln * Cn + c;
    float s = 0.f;
    for (int l = 0; l < Ln; ++l) s += p[(long)l * Cn];
    out[i] = s / (float)Ln;
}

// trend_init = [t_init[:, -512:], mean], seasonal_init = [s_init[:, -512:], 0]
__global__ void k_build_inits(const float* __restrict__ sI, const float* __restrict__ tI,
                              const float* __restrict__ xmean,
                              float* __restrict__ seas, float* __restrict__ tr) {
    long i = (long)blockIdx.x * blockDim.x + threadIdx.x;
    if (i >= 8L * 1024 * 64) return;
    int  c = (int)(i & 63);
    long t = i >> 6;
    int  l = (int)(t & 1023);
    int  b = (int)(t >> 10);
    if (l < 512) {
        long src = ((long)b * 1024 + 512 + l) * 64 + c;
        seas[i] = sI[src];
        tr[i]   = tI[src];
    } else {
        seas[i] = 0.f;
        tr[i]   = xmean[b * 64 + c];
    }
}

__global__ __launch_bounds__(256)
void k_layernorm(const float* __restrict__ x, const float* __restrict__ g,
                 const float* __restrict__ bb, float* __restrict__ y, int Cn) {
    __shared__ float red[256];
    int row = blockIdx.x;
    const float* xr = x + (long)row * Cn;
    float s = 0.f;
    for (int c = threadIdx.x; c < Cn; c += 256) s += xr[c];
    red[threadIdx.x] = s; __syncthreads();
    for (int o = 128; o > 0; o >>= 1) {
        if (threadIdx.x < o) red[threadIdx.x] += red[threadIdx.x + o];
        __syncthreads();
    }
    float mean = red[0] / (float)Cn; __syncthreads();
    float v = 0.f;
    for (int c = threadIdx.x; c < Cn; c += 256) { float d = xr[c] - mean; v += d * d; }
    red[threadIdx.x] = v; __syncthreads();
    for (int o = 128; o > 0; o >>= 1) {
        if (threadIdx.x < o) red[threadIdx.x] += red[threadIdx.x + o];
        __syncthreads();
    }
    float inv = rsqrtf(red[0] / (float)Cn + 1e-5f);
    float* yr = y + (long)row * Cn;
    for (int c = threadIdx.x; c < Cn; c += 256)
        yr[c] = (xr[c] - mean) * inv * g[c] + bb[c];
}

// mean_corr[b,tau] = (1/512) * sum_l S[b, (l+tau)%1024, l]   (S = Q K^T)
__global__ void k_corr_diag(const float* __restrict__ S, float* __restrict__ corr) {
    int b   = blockIdx.x;
    int tau = blockIdx.y * blockDim.x + threadIdx.x;
    const float* Sb = S + (long)b * 1024 * 1024;
    float s = 0.f;
    for (int l = 0; l < 1024; ++l)
        s += Sb[(long)((l + tau) & 1023) * 1024 + l];
    corr[b * 1024 + tau] = s * (1.0f / 512.0f);
}

// top-20 delays of batch-mean correlation (iterative argmax; deterministic)
__global__ __launch_bounds__(1024)
void k_topk(const float* __restrict__ corr, int* __restrict__ delays) {
    __shared__ float g[1024];
    int t = threadIdx.x;
    float s = 0.f;
    for (int b = 0; b < 8; ++b) s += corr[b * 1024 + t];
    g[t] = s * 0.125f;
    __syncthreads();
    if (t == 0) {
        for (int k = 0; k < 20; ++k) {
            int best = 0; float bv = g[0];
            for (int i = 1; i < 1024; ++i) if (g[i] > bv) { bv = g[i]; best = i; }
            delays[k] = best;
            g[best] = -3.0e38f;
        }
    }
}

__global__ void k_softmax_sel(const float* __restrict__ corr, const int* __restrict__ delays,
                              float* __restrict__ w) {
    int b = blockIdx.x;
    if (threadIdx.x == 0) {
        float vals[20]; float mx = -3.0e38f;
        for (int k = 0; k < 20; ++k) {
            vals[k] = corr[b * 1024 + delays[k]];
            mx = vals[k] > mx ? vals[k] : mx;
        }
        float s = 0.f;
        for (int k = 0; k < 20; ++k) { vals[k] = expf(vals[k] - mx); s += vals[k]; }
        for (int k = 0; k < 20; ++k) w[b * 20 + k] = vals[k] / s;
    }
}

// out[b,l,c] = sum_k w[b,k] * V[b,(l+delay_k)%1024,c]
__global__ __launch_bounds__(256)
void k_rollsum(const float* __restrict__ V, const float* __restrict__ w,
               const int* __restrict__ delays, float* __restrict__ out) {
    __shared__ int   sd[20];
    __shared__ float sw[20];
    long i = (long)blockIdx.x * 256 + threadIdx.x;   // over 8*1024*512
    int  b = (int)(i >> 19);
    if (threadIdx.x < 20) {
        sd[threadIdx.x] = delays[threadIdx.x];
        sw[threadIdx.x] = w[b * 20 + threadIdx.x];
    }
    __syncthreads();
    int c = (int)(i & 511);
    int l = (int)((i >> 9) & 1023);
    const float* Vb = V + (long)b * 1024 * 512;
    float s = 0.f;
    for (int k = 0; k < 20; ++k) {
        int ls = (l + sd[k]) & 1023;
        s += sw[k] * Vb[(long)ls * 512 + c];
    }
    out[i] = s;
}

__global__ void k_add(const float* __restrict__ a, const float* __restrict__ b2,
                      float* __restrict__ c, long n) {
    long i = (long)blockIdx.x * blockDim.x + threadIdx.x;
    if (i < n) c[i] = a[i] + b2[i];
}

__global__ void k_slice_out(const float* __restrict__ s, float* __restrict__ d) {
    long i = (long)blockIdx.x * blockDim.x + threadIdx.x;   // over 8*512*64
    if (i >= 8L * 512 * 64) return;
    int  c = (int)(i & 63);
    long t = i >> 6;
    int  l = (int)(t & 511);
    int  b = (int)(t >> 9);
    d[i] = s[((long)b * 1024 + 512 + l) * 64 + c];
}

// ---------------------------------------------------------------------------
// Host orchestration
// ---------------------------------------------------------------------------
extern "C" void kernel_launch(void* const* d_in, const int* in_sizes, int n_in,
                              void* d_out, int out_size, void* d_ws, size_t ws_size,
                              hipStream_t stream)
{
    (void)in_sizes; (void)n_in; (void)out_size; (void)ws_size;

    // ---- inputs (insertion-order flatten of setup_inputs) ----
    const float* x          = (const float*)d_in[0];
    const float* w_emb      = (const float*)d_in[1];
    const float* b_emb      = (const float*)d_in[2];
    const float* enc_ac[4]  = {(const float*)d_in[3], (const float*)d_in[4],
                               (const float*)d_in[5], (const float*)d_in[6]};
    const float* enc_conv1  = (const float*)d_in[7];
    const float* enc_conv2  = (const float*)d_in[8];
    const float* enc_ln_g   = (const float*)d_in[9];
    const float* enc_ln_b   = (const float*)d_in[10];
    const float* dec_ac1[4] = {(const float*)d_in[11], (const float*)d_in[12],
                               (const float*)d_in[13], (const float*)d_in[14]};
    const float* dec_ac2[4] = {(const float*)d_in[15], (const float*)d_in[16],
                               (const float*)d_in[17], (const float*)d_in[18]};
    const float* lin1       = (const float*)d_in[19];
    const float* lin2       = (const float*)d_in[20];
    const float* lin3       = (const float*)d_in[21];
    const float* dec_conv1  = (const float*)d_in[22];
    const float* dec_conv2  = (const float*)d_in[23];
    const float* lin_season = (const float*)d_in[24];
    const float* dec_ln_g   = (const float*)d_in[25];
    const float* dec_ln_b   = (const float*)d_in[26];
    const float* w_final    = (const float*)d_in[27];
    const float* b_final    = (const float*)d_in[28];
    float* out = (float*)d_out;

    // ---- workspace carve ----
    char*  wsb = (char*)d_ws;
    size_t off = 0;
    auto carve = [&](size_t bytes) -> void* {
        void* r = wsb + off;
        off += (bytes + 255) & ~(size_t)255;
        return r;
    };
    bfloat* wT_emb = (bfloat*)carve(512L * 64 * 2);
    bfloat* encT[4];  for (int i = 0; i < 4; ++i) encT[i]  = (bfloat*)carve(512L * 512 * 2);
    bfloat* dec1T[4]; for (int i = 0; i < 4; ++i) dec1T[i] = (bfloat*)carve(512L * 512 * 2);
    bfloat* dec2T[4]; for (int i = 0; i < 4; ++i) dec2T[i] = (bfloat*)carve(512L * 512 * 2);
    bfloat* lin1T     = (bfloat*)carve(512L * 512 * 2);
    bfloat* lin2T     = (bfloat*)carve(512L * 512 * 2);
    bfloat* lin3T     = (bfloat*)carve(512L * 512 * 2);
    bfloat* linSeasT  = (bfloat*)carve(512L * 512 * 2);
    bfloat* wfinT     = (bfloat*)carve(64L * 512 * 2);
    bfloat* enc_c1T   = (bfloat*)carve(3L * 2048 * 512 * 2);
    bfloat* enc_c2T   = (bfloat*)carve(3L * 512 * 2048 * 2);
    bfloat* dec_c1T   = (bfloat*)carve(3L * 2048 * 512 * 2);
    bfloat* dec_c2T   = (bfloat*)carve(3L * 512 * 2048 * 2);
    bfloat* H1 = (bfloat*)carve(8192L * 512 * 2);
    bfloat* H2 = (bfloat*)carve(8192L * 512 * 2);
    bfloat* HC = (bfloat*)carve(8192L * 2048 * 2);
    float* F1 = (float*)carve(8192L * 512 * 4);
    float* F2 = (float*)carve(8192L * 512 * 4);
    float* F3 = (float*)carve(8192L * 512 * 4);
    float* F4 = (float*)carve(8192L * 512 * 4);
    float* F5 = (float*)carve(8192L * 512 * 4);
    float* FC = (float*)carve(8192L * 2048 * 4);
    float* FS = (float*)carve(8L * 1024 * 1024 * 4);
    float* sInit   = (float*)carve(8L * 1024 * 64 * 4);
    float* tInit   = (float*)carve(8L * 1024 * 64 * 4);
    float* seasIn  = (float*)carve(8L * 1024 * 64 * 4);
    float* trendIn = (float*)carve(8L * 1024 * 64 * 4);
    float* xmean   = (float*)carve(8L * 64 * 4);
    float* corr    = (float*)carve(8L * 1024 * 4);
    int*   delays  = (int*)carve(32 * 4);
    float* wsoft   = (float*)carve(8L * 32 * 4);

    // ---- launch helpers ----
    auto CV = [&](const float* s, bfloat* d, long n) {
        k_f32_to_bf16<<<dim3((unsigned)((n + 255) / 256)), dim3(256), 0, stream>>>(s, d, n);
    };
    auto TW = [&](const float* W, bfloat* Wt, int K, int N) {
        long n = (long)K * N;
        k_transpose_bf16<<<dim3((unsigned)((n + 255) / 256)), dim3(256), 0, stream>>>(W, Wt, K, N);
    };
    auto CW = [&](const float* W, bfloat* Wt, int O, int I) {
        long n = 3L * O * I;
        k_convw_tap<<<dim3((unsigned)((n + 255) / 256)), dim3(256), 0, stream>>>(W, Wt, O, I);
    };
    auto GEMM = [&](const bfloat* A, const bfloat* Bt, float* C, const float* bias,
                    const float* resid, int M, int N, int K, int shift, int flags,
                    int batch, long sA, long sB, long sC) {
        dim3 g(N / 32, M / 64, batch);
        k_wmma_gemm<<<g, dim3(256), 0, stream>>>(A, Bt, C, bias, resid,
                                                 M, N, K, 1024, shift, flags, sA, sB, sC);
    };
    auto DECOMP = [&](const float* X, float* S, float* Tr, int C) {
        long n = 8L * 1024 * C;
        k_decomp<<<dim3((unsigned)((n + 255) / 256)), dim3(256), 0, stream>>>(X, S, Tr, 8, 1024, C);
    };
    // conv1d(k=3, replicate pad) as 3 accumulating GEMM taps; LEAKY + resid on last
    auto CONV3 = [&](const float* X, bfloat* Xh, const bfloat* Wt, float* Y,
                     int Kc, int Nc, const float* resid) {
        CV(X, Xh, 8192L * Kc);
        for (int t = 0; t < 3; ++t) {
            int fl = (t == 0 ? 0 : GF_ACC) | (t == 2 ? GF_LEAKY : 0);
            GEMM(Xh, Wt + (long)t * Nc * Kc, Y, nullptr,
                 (t == 2 ? resid : nullptr), 8192, Nc, Kc, t - 1, fl, 1, 0, 0, 0);
        }
    };
    // AutoCorrelation block: OUT = roll_agg(V) @ Wo + Xq
    auto AUTOC = [&](const float* Xq, const float* Xkv, const bfloat* const* WT,
                     float* FQ, float* FK, float* FV, float* OUT) {
        CV(Xq, H1, 8192L * 512);
        const bfloat* Hkv = H1;
        if (Xkv != Xq) { CV(Xkv, H2, 8192L * 512); Hkv = H2; }
        GEMM(H1,  WT[0], FQ, nullptr, nullptr, 8192, 512, 512, 0, 0, 1, 0, 0, 0);
        GEMM(Hkv, WT[1], FK, nullptr, nullptr, 8192, 512, 512, 0, 0, 1, 0, 0, 0);
        GEMM(Hkv, WT[2], FV, nullptr, nullptr, 8192, 512, 512, 0, 0, 1, 0, 0, 0);
        CV(FQ, H1, 8192L * 512);
        CV(FK, H2, 8192L * 512);
        // batched S[b] = Qb @ Kb^T  (K rows are already [N,K] = Bt layout)
        GEMM(H1, H2, FS, nullptr, nullptr, 1024, 1024, 512, 0, 0,
             8, 1024L * 512, 1024L * 512, 1024L * 1024);
        k_corr_diag<<<dim3(8, 4), dim3(256), 0, stream>>>(FS, corr);
        k_topk<<<dim3(1), dim3(1024), 0, stream>>>(corr, delays);
        k_softmax_sel<<<dim3(8), dim3(32), 0, stream>>>(corr, delays, wsoft);
        k_rollsum<<<dim3(16384), dim3(256), 0, stream>>>(FV, wsoft, delays, FQ);
        CV(FQ, H1, 8192L * 512);
        GEMM(H1, WT[3], OUT, nullptr, Xq, 8192, 512, 512, 0, 0, 1, 0, 0, 0);
    };

    // ---- weight prep (bf16, [N,K] layout; conv weights tap-major) ----
    TW(w_emb, wT_emb, 64, 512);
    for (int i = 0; i < 4; ++i) TW(enc_ac[i],  encT[i],  512, 512);
    for (int i = 0; i < 4; ++i) TW(dec_ac1[i], dec1T[i], 512, 512);
    for (int i = 0; i < 4; ++i) TW(dec_ac2[i], dec2T[i], 512, 512);
    TW(lin1, lin1T, 512, 512);
    TW(lin2, lin2T, 512, 512);
    TW(lin3, lin3T, 512, 512);
    TW(lin_season, linSeasT, 512, 512);
    TW(w_final, wfinT, 512, 64);
    CW(enc_conv1, enc_c1T, 2048, 512);
    CW(enc_conv2, enc_c2T, 512, 2048);
    CW(dec_conv1, dec_c1T, 2048, 512);
    CW(dec_conv2, dec_c2T, 512, 2048);

    // ---- input decomposition / inits ----
    DECOMP(x, sInit, tInit, 64);
    k_mean_l<<<dim3(2), dim3(256), 0, stream>>>(x, xmean, 8, 1024, 64);
    k_build_inits<<<dim3((unsigned)((8L * 1024 * 64 + 255) / 256)), dim3(256), 0, stream>>>(
        sInit, tInit, xmean, seasIn, trendIn);

    // ---- encoder ----
    CV(x, H1, 8192L * 64);
    GEMM(H1, wT_emb, F1, b_emb, nullptr, 8192, 512, 64, 0, 0, 1, 0, 0, 0);   // emb(x)
    AUTOC(F1, F1, encT, F2, F3, F4, F5);                                     // a = ac+x
    DECOMP(F5, F2, F3, 512);                                                 // s1=F2
    CONV3(F2, H1, enc_c1T, FC, 512, 2048, nullptr);                          // leaky(conv1)
    CONV3(FC, HC, enc_c2T, F3, 2048, 512, F2);                               // leaky(conv2)+s1
    DECOMP(F3, F4, F5, 512);                                                 // season=F4
    k_layernorm<<<dim3(8192), dim3(256), 0, stream>>>(F4, enc_ln_g, enc_ln_b, F1, 512); // enc_out=F1

    // ---- decoder ----
    CV(seasIn, H1, 8192L * 64);
    GEMM(H1, wT_emb, F2, b_emb, nullptr, 8192, 512, 64, 0, 0, 1, 0, 0, 0);   // dS=F2
    CV(trendIn, H1, 8192L * 64);
    GEMM(H1, wT_emb, F3, b_emb, nullptr, 8192, 512, 64, 0, 0, 1, 0, 0, 0);   // trend acc=F3

    AUTOC(F2, F2, dec1T, F4, F5, FC, F5);                                    // a1=F5
    DECOMP(F5, F4, F2, 512);                                                 // s1=F4, t1=F2
    CV(F2, H1, 8192L * 512);
    GEMM(H1, lin1T, F3, nullptr, nullptr, 8192, 512, 512, 0, GF_ACC, 1, 0, 0, 0);

    AUTOC(F4, F1, dec2T, F2, F5, FC, F5);                                    // a2=F5
    DECOMP(F5, F2, F4, 512);                                                 // s2=F2, t2=F4
    CV(F4, H1, 8192L * 512);
    GEMM(H1, lin2T, F3, nullptr, nullptr, 8192, 512, 512, 0, GF_ACC, 1, 0, 0, 0);

    CONV3(F2, H1, dec_c1T, FC, 512, 2048, nullptr);                          // leaky(conv1)
    CONV3(FC, HC, dec_c2T, F4, 2048, 512, F2);                               // leaky(conv2)+s2
    DECOMP(F4, F5, F2, 512);                                                 // s3=F5, t3=F2
    CV(F2, H1, 8192L * 512);
    GEMM(H1, lin3T, F3, nullptr, nullptr, 8192, 512, 512, 0, GF_ACC, 1, 0, 0, 0);

    k_layernorm<<<dim3(8192), dim3(256), 0, stream>>>(F5, dec_ln_g, dec_ln_b, F2, 512);
    CV(F2, H1, 8192L * 512);
    GEMM(H1, linSeasT, F4, nullptr, nullptr, 8192, 512, 512, 0, 0, 1, 0, 0, 0); // season=F4

    // ---- head: (season + trend) @ w_final + b_final, slice last 512 steps ----
    k_add<<<dim3((unsigned)((8192L * 512 + 255) / 256)), dim3(256), 0, stream>>>(
        F4, F3, F5, 8192L * 512);
    CV(F5, H1, 8192L * 512);
    GEMM(H1, wfinT, F2, b_final, nullptr, 8192, 64, 512, 0, 0, 1, 0, 0, 0);
    k_slice_out<<<dim3((unsigned)((8L * 512 * 64 + 255) / 256)), dim3(256), 0, stream>>>(F2, out);
}
